// MultiHeadAttention_75685913690598
// MI455X (gfx1250) — matrix operand order repaired and checked by reading, not compiled, not run on hardware
//
#include <hip/hip_runtime.h>

// ---------------------------------------------------------------------------
// MI455X (gfx1250, wave32) multi-head attention:
//   Q/K/V projections (bf16 WMMA GEMM) -> causal flash attention (bf16 WMMA)
//   -> output projection (bf16 WMMA GEMM, fp32 out).
// Uses v_wmma_f32_16x16x32_bf16 for all matmuls; async global->LDS copies
// (ASYNCcnt path) where tiles need no transpose; global_prefetch for W tiles.
// ---------------------------------------------------------------------------

typedef __attribute__((ext_vector_type(16))) __bf16 v16bf;
typedef __attribute__((ext_vector_type(8)))  float  v8f;
typedef int v4i __attribute__((vector_size(16)));

#define D_MODEL 1024
#define SEQ     2048
#define NH      16
#define DK      64

#if __has_builtin(__builtin_amdgcn_global_load_async_to_lds_b128)
#define HAVE_ASYNC_LDS 1
#endif

// Copy 16 bytes global -> LDS. Async (ASYNCcnt-tracked) when available.
__device__ __forceinline__ void copy_b128_to_lds(void* lds, const void* g) {
#ifdef HAVE_ASYNC_LDS
  __builtin_amdgcn_global_load_async_to_lds_b128(
      (__attribute__((address_space(1))) v4i*)(void*)(size_t)g,
      (__attribute__((address_space(3))) v4i*)lds, 0, 0);
#else
  *(uint4*)lds = *(const uint4*)g;
#endif
}
__device__ __forceinline__ void async_wait_all() {
#ifdef HAVE_ASYNC_LDS
  asm volatile("s_wait_asynccnt 0x0" ::: "memory");
#endif
}

__device__ __forceinline__ unsigned short f2bf(float f) {
  union { float f; unsigned u; } v; v.f = f;
  unsigned r = v.u + 0x7FFFu + ((v.u >> 16) & 1u);   // round-to-nearest-even
  return (unsigned short)(r >> 16);
}

__device__ __forceinline__ v8f zero8() {
  v8f z;
#pragma unroll
  for (int i = 0; i < 8; ++i) z[i] = 0.0f;
  return z;
}

union FragU { uint4 u[2]; v16bf v; };

// A-fragment (16x32 bf16, M=lane&15):
//   lanes 0-15 : elems 0..7 -> K=0..7,  elems 8..15 -> K=16..23
//   lanes 16-31: elems 0..7 -> K=8..15, elems 8..15 -> K=24..31
__device__ __forceinline__ v16bf load_afrag(const unsigned short* base, int ld, int lane) {
  const int r    = lane & 15;
  const int koff = (lane >> 4) << 3;
  const unsigned short* p = base + r * ld + koff;
  FragU f;
  f.u[0] = *(const uint4*)(p);
  f.u[1] = *(const uint4*)(p + 16);
  return f.v;
}

// B-fragment (32x16 bf16, N=lane&15): lanes 0-15 K=0..7,16..23; lanes 16-31 K=8..15,24..31.
// LDS tile stored K-contiguous per N row ("[n][k]").
__device__ __forceinline__ v16bf load_bfrag(const unsigned short* base, int ld, int lane) {
  const int n  = lane & 15;
  const int kb = (lane >> 4) << 3;
  const unsigned short* p = base + n * ld + kb;
  FragU f;
  f.u[0] = *(const uint4*)(p);
  f.u[1] = *(const uint4*)(p + 16);
  return f.v;
}

__device__ __forceinline__ v8f wmma_bf16(v16bf a, v16bf b, v8f c) {
  return __builtin_amdgcn_wmma_f32_16x16x32_bf16(false, a, false, b, (short)0, c, false, false);
}

// ---------------------------------------------------------------------------
// GEMM: C[M,N] = A[M,K] @ W[K,N] + bias.  A fp32 or bf16, W fp32 (transposed
// to bf16 in LDS), C bf16 or fp32.  BM=128 BN=128 BK=32, 256 threads (8 waves),
// each wave computes 32x64 via 2x4 WMMA fragments.
// ---------------------------------------------------------------------------
template <bool A_F32, bool OUT_BF16>
__global__ __launch_bounds__(256) void mha_gemm_kernel(
    const void* __restrict__ Aptr, const float* __restrict__ W,
    const float* __restrict__ bias, void* __restrict__ Cptr,
    int M, int N, int K) {
  __shared__ unsigned short As[128 * 32];
  __shared__ unsigned short Bs[128 * 32];

  const int tid  = threadIdx.x;
  const int lane = tid & 31;
  const int w    = tid >> 5;
  const int wm   = (w & 3) * 32;
  const int wn   = (w >> 2) * 64;
  const int m0   = blockIdx.y * 128;
  const int n0   = blockIdx.x * 128;

  v8f acc[2][4];
#pragma unroll
  for (int i = 0; i < 2; ++i)
#pragma unroll
    for (int j = 0; j < 4; ++j) acc[i][j] = zero8();

  const int aRow = tid >> 1;        // 0..127
  const int aCol = (tid & 1) * 16;  // 0 / 16
  const int bRow = tid >> 3;        // 0..31  (k)
  const int bCol = (tid & 7) * 16;  // 0..112 (n)

  for (int k0 = 0; k0 < K; k0 += 32) {
    __syncthreads();
    // ---- prefetch next tiles into L2 while we fill/consume this one ----
    if (k0 + 32 < K) {
      __builtin_prefetch(W + (size_t)(k0 + 32 + bRow) * N + n0 + bCol, 0, 3);
      if constexpr (A_F32)
        __builtin_prefetch((const float*)Aptr + (size_t)(m0 + aRow) * K + k0 + 32 + aCol, 0, 3);
    }
    // ---- load A tile (row-major [m][k], bf16) ----
    if constexpr (A_F32) {
      const float* A = (const float*)Aptr;
      const float* src = A + (size_t)(m0 + aRow) * K + k0 + aCol;
      union { unsigned short s[16]; uint4 u[2]; } tmp;
#pragma unroll
      for (int j = 0; j < 16; j += 4) {
        float4 f = *(const float4*)(src + j);
        tmp.s[j + 0] = f2bf(f.x); tmp.s[j + 1] = f2bf(f.y);
        tmp.s[j + 2] = f2bf(f.z); tmp.s[j + 3] = f2bf(f.w);
      }
      uint4* dst = (uint4*)&As[aRow * 32 + aCol];
      dst[0] = tmp.u[0]; dst[1] = tmp.u[1];
    } else {
      const unsigned short* A = (const unsigned short*)Aptr;
      const unsigned short* src = A + (size_t)(m0 + aRow) * K + k0 + aCol;
      copy_b128_to_lds(&As[aRow * 32 + aCol + 0], src + 0);
      copy_b128_to_lds(&As[aRow * 32 + aCol + 8], src + 8);
    }
    // ---- load W tile, transpose into Bs[n][k] (bf16) ----
    {
      const float* src = W + (size_t)(k0 + bRow) * N + n0 + bCol;
#pragma unroll
      for (int j = 0; j < 16; j += 4) {
        float4 f = *(const float4*)(src + j);
        Bs[(bCol + j + 0) * 32 + bRow] = f2bf(f.x);
        Bs[(bCol + j + 1) * 32 + bRow] = f2bf(f.y);
        Bs[(bCol + j + 2) * 32 + bRow] = f2bf(f.z);
        Bs[(bCol + j + 3) * 32 + bRow] = f2bf(f.w);
      }
    }
    async_wait_all();
    __syncthreads();
    // ---- compute: one WMMA K-step (BK==32) ----
    v16bf a0 = load_afrag(As + (wm + 0) * 32, 32, lane);
    v16bf a1 = load_afrag(As + (wm + 16) * 32, 32, lane);
#pragma unroll
    for (int ns = 0; ns < 4; ++ns) {
      v16bf b = load_bfrag(Bs + (wn + ns * 16) * 32, 32, lane);
      acc[0][ns] = wmma_bf16(a0, b, acc[0][ns]);
      acc[1][ns] = wmma_bf16(a1, b, acc[1][ns]);
    }
  }

  // ---- epilogue: + bias, store ----
  const int ncol = lane & 15;
  const int mAdd = (lane < 16) ? 0 : 8;
#pragma unroll
  for (int ms = 0; ms < 2; ++ms) {
#pragma unroll
    for (int ns = 0; ns < 4; ++ns) {
      const int col = n0 + wn + ns * 16 + ncol;
      const float bv = bias[col];
#pragma unroll
      for (int i = 0; i < 8; ++i) {
        const int row = m0 + wm + ms * 16 + mAdd + i;
        const float v = acc[ms][ns][i] + bv;
        if constexpr (OUT_BF16)
          ((unsigned short*)Cptr)[(size_t)row * N + col] = f2bf(v);
        else
          ((float*)Cptr)[(size_t)row * N + col] = v;
      }
    }
  }
}

// ---------------------------------------------------------------------------
// Causal flash attention.  Grid: (SEQ/128, NH, BATCH), 256 threads (8 waves).
// Each wave owns 16 queries x full Dk=64.  K/V streamed in 64-key LDS tiles.
// ---------------------------------------------------------------------------
__global__ __launch_bounds__(256) void mha_attn_kernel(
    const unsigned short* __restrict__ Q, const unsigned short* __restrict__ Kg,
    const unsigned short* __restrict__ Vg, unsigned short* __restrict__ O) {
  __shared__ unsigned short Qs[128 * 64];       // [q][d]
  __shared__ unsigned short Ks[64 * 64];        // [key][d]  (B-frag: N=key, K=d)
  __shared__ unsigned short Vt[64 * 64];        // [d][key]  (B-frag: N=d,   K=key)
  __shared__ unsigned short Ps[8][16 * 64];     // per-wave P tile [q][key]

  const int tid  = threadIdx.x;
  const int lane = tid & 31;
  const int w    = tid >> 5;
  const int qt0  = blockIdx.x * 128;
  const int h    = blockIdx.y;
  const int b    = blockIdx.z;
  const size_t rowG = (size_t)b * SEQ;
  const int col0 = h * DK;

  // load Q tile: 128x64 bf16 (async global->LDS)
#pragma unroll
  for (int i = 0; i < 4; ++i) {
    const int idx = tid + 256 * i;
    const int r = idx >> 3, c = (idx & 7) * 8;
    copy_b128_to_lds(&Qs[r * 64 + c],
                     &Q[(rowG + qt0 + r) * D_MODEL + col0 + c]);
  }

  v8f acc[4];
#pragma unroll
  for (int ns = 0; ns < 4; ++ns) acc[ns] = zero8();
  float mrow[8], lrow[8];
#pragma unroll
  for (int i = 0; i < 8; ++i) { mrow[i] = -1e30f; lrow[i] = 0.0f; }

  const int ncol = lane & 15;
  const int mAdd = (lane < 16) ? 0 : 8;
  const int qb_w = qt0 + w * 16;

  const int nkt = 2 * blockIdx.x + 2;  // causal: key tiles 0 .. max query tile
  for (int kt = 0; kt < nkt; ++kt) {
    const int kb = kt * 64;
    __syncthreads();
    // K tile (row-major, async global->LDS)
#pragma unroll
    for (int i = 0; i < 2; ++i) {
      const int idx = tid + 256 * i;
      const int r = idx >> 3, c = (idx & 7) * 8;
      copy_b128_to_lds(&Ks[r * 64 + c],
                       &Kg[(rowG + kb + r) * D_MODEL + col0 + c]);
    }
    // V tile, transposed into Vt[d][key]
    {
      const int r  = tid >> 2;        // key 0..63
      const int cs = (tid & 3) * 16;  // d start
      const unsigned short* src = &Vg[(rowG + kb + r) * D_MODEL + col0 + cs];
      union { unsigned short s[16]; uint4 u[2]; } tmp;
      tmp.u[0] = *(const uint4*)(src);
      tmp.u[1] = *(const uint4*)(src + 8);
#pragma unroll
      for (int j = 0; j < 16; ++j) Vt[(cs + j) * 64 + r] = tmp.s[j];
    }
    async_wait_all();
    __syncthreads();

    // ---- scores S = Q @ K^T ----
    v8f sc[4];
#pragma unroll
    for (int ns = 0; ns < 4; ++ns) sc[ns] = zero8();
#pragma unroll
    for (int ks = 0; ks < 2; ++ks) {
      v16bf aq = load_afrag(Qs + (w * 16) * 64 + ks * 32, 64, lane);
#pragma unroll
      for (int ns = 0; ns < 4; ++ns) {
        v16bf bk = load_bfrag(Ks + (ns * 16) * 64 + ks * 32, 64, lane);
        sc[ns] = wmma_bf16(aq, bk, sc[ns]);
      }
    }

    // ---- scale + causal mask ----
    const bool needMask = (kb + 63) > qb_w;
#pragma unroll
    for (int ns = 0; ns < 4; ++ns) {
      const int key = kb + ns * 16 + ncol;
#pragma unroll
      for (int i = 0; i < 8; ++i) {
        float s = sc[ns][i] * 0.125f;  // 1/sqrt(64)
        if (needMask && key > (qb_w + mAdd + i)) s = -1e30f;
        sc[ns][i] = s;
      }
    }

    // ---- online softmax (row reductions across 16-lane halves) ----
#pragma unroll
    for (int i = 0; i < 8; ++i) {
      float mx = fmaxf(fmaxf(sc[0][i], sc[1][i]), fmaxf(sc[2][i], sc[3][i]));
      mx = fmaxf(mx, __shfl_xor(mx, 1, 32));
      mx = fmaxf(mx, __shfl_xor(mx, 2, 32));
      mx = fmaxf(mx, __shfl_xor(mx, 4, 32));
      mx = fmaxf(mx, __shfl_xor(mx, 8, 32));
      const float mn   = fmaxf(mrow[i], mx);
      const float corr = __expf(mrow[i] - mn);
      mrow[i] = mn;
      float ps = 0.0f;
#pragma unroll
      for (int ns = 0; ns < 4; ++ns) {
        const float p = __expf(sc[ns][i] - mn);
        sc[ns][i] = p;
        ps += p;
      }
      ps += __shfl_xor(ps, 1, 32);
      ps += __shfl_xor(ps, 2, 32);
      ps += __shfl_xor(ps, 4, 32);
      ps += __shfl_xor(ps, 8, 32);
      lrow[i] = lrow[i] * corr + ps;
#pragma unroll
      for (int ns = 0; ns < 4; ++ns) acc[ns][i] *= corr;
    }

    // ---- repack P (C-layout -> A-layout) through per-wave LDS ----
#pragma unroll
    for (int ns = 0; ns < 4; ++ns)
#pragma unroll
      for (int i = 0; i < 8; ++i)
        Ps[w][(mAdd + i) * 64 + ns * 16 + ncol] = f2bf(sc[ns][i]);
    asm volatile("s_wait_dscnt 0x0" ::: "memory");  // per-wave LDS RAW

    // ---- acc += P @ V ----
#pragma unroll
    for (int ks = 0; ks < 2; ++ks) {
      v16bf ap = load_afrag(&Ps[w][ks * 32], 64, lane);
#pragma unroll
      for (int ns = 0; ns < 4; ++ns) {
        v16bf bv = load_bfrag(Vt + (ns * 16) * 64 + ks * 32, 64, lane);
        acc[ns] = wmma_bf16(ap, bv, acc[ns]);
      }
    }
  }

  // ---- normalize and store O (bf16) ----
#pragma unroll
  for (int ns = 0; ns < 4; ++ns) {
    const int col = col0 + ns * 16 + ncol;
#pragma unroll
    for (int i = 0; i < 8; ++i) {
      const size_t row = rowG + qb_w + mAdd + i;
      const float v = acc[ns][i] / lrow[i];
      O[row * D_MODEL + col] = f2bf(v);
    }
  }
}

// ---------------------------------------------------------------------------
extern "C" void kernel_launch(void* const* d_in, const int* in_sizes, int n_in,
                              void* d_out, int out_size, void* d_ws, size_t ws_size,
                              hipStream_t stream) {
  (void)in_sizes; (void)n_in; (void)out_size; (void)ws_size;
  const float* X   = (const float*)d_in[0];   // context  [B*S, D]
  const float* Vin = (const float*)d_in[1];   // value    [B*S, D]
  // d_in[2] = mask (causal; computed analytically)
  const float* Wq = (const float*)d_in[3];
  const float* bq = (const float*)d_in[4];
  const float* Wk = (const float*)d_in[5];
  const float* bk = (const float*)d_in[6];
  const float* Wv = (const float*)d_in[7];
  const float* bv = (const float*)d_in[8];
  const float* Wo = (const float*)d_in[9];
  const float* bo = (const float*)d_in[10];

  const int M = 4 * SEQ;           // 8192
  const size_t nElem = (size_t)M * D_MODEL;

  unsigned short* Qb = (unsigned short*)d_ws;
  unsigned short* Kb = Qb + nElem;
  unsigned short* Vb = Kb + nElem;
  unsigned short* Ob = Vb + nElem;   // total 4 * 16.8 MB = 67 MB bf16 scratch

  dim3 gGrid(D_MODEL / 128, M / 128);  // (8, 64)
  dim3 blk(256);

  mha_gemm_kernel<true, true><<<gGrid, blk, 0, stream>>>(X,   Wq, bq, Qb, M, D_MODEL, D_MODEL);
  mha_gemm_kernel<true, true><<<gGrid, blk, 0, stream>>>(X,   Wk, bk, Kb, M, D_MODEL, D_MODEL);
  mha_gemm_kernel<true, true><<<gGrid, blk, 0, stream>>>(Vin, Wv, bv, Vb, M, D_MODEL, D_MODEL);

  dim3 aGrid(SEQ / 128, NH, 4);        // (16, 16, 4)
  mha_attn_kernel<<<aGrid, blk, 0, stream>>>(Qb, Kb, Vb, Ob);

  mha_gemm_kernel<false, false><<<gGrid, blk, 0, stream>>>(Ob, Wo, bo, (float*)d_out, M, D_MODEL, D_MODEL);
}
